// MolecularGraphNeuralNetwork_5523327943095
// MI455X (gfx1250) — matrix operand order repaired and checked by reading, not compiled, not run on hardware
//
#include <hip/hip_runtime.h>

typedef __attribute__((ext_vector_type(16))) _Float16 v16h;
typedef __attribute__((ext_vector_type(8)))  float    v8f;

#define GN      100000
#define GE      1600000
#define GG      2048
#define GD      128
#define GF_IN   9
#define BN_EPS  1e-5f
#define WLDS_STRIDE 136   // halves; pad to break LDS bank alignment

__device__ __forceinline__ void atomicAddF(float* p, float v) {
    __hip_atomic_fetch_add(p, v, __ATOMIC_RELAXED, __HIP_MEMORY_SCOPE_AGENT);
}

// ---------------------------------------------------------------- utilities
__global__ void k_fill(float* __restrict__ p, float v, int n) {
    int i = blockIdx.x * blockDim.x + threadIdx.x;
    if (i < n) p[i] = v;
}

__global__ void k_count_deg(const long long* __restrict__ ei, float* __restrict__ deg, int ne) {
    int e = blockIdx.x * blockDim.x + threadIdx.x;
    if (e < ne) atomicAddF(&deg[(int)ei[(size_t)ne + e]], 1.0f);
}

__global__ void k_dinv(const float* __restrict__ deg, float* __restrict__ dinv, int n) {
    int i = blockIdx.x * blockDim.x + threadIdx.x;
    if (i < n) dinv[i] = rsqrtf(deg[i]);
}

// ---------------------------------------------------------------- WMMA GEMM
// lin = affine(A) @ W        (affine = per-K BN scale/shift, compile-time toggle)
// agg = dinv^2 * lin + bias  (self-loop init for the edge scatter)
// A: [N x K] row-major f32, W: [K x 128] row-major f32. N % 16 == 0.
// K is a compile-time constant -> no dynamic guards, vectorized A loads.
template<int K, bool AFFINE>
__global__ void k_gemm_wmma(const float* __restrict__ A,
                            const float* __restrict__ W,
                            const float* __restrict__ bias,
                            const float* __restrict__ scale,
                            const float* __restrict__ shift,
                            const float* __restrict__ dinv,
                            float* __restrict__ lin,
                            float* __restrict__ agg,
                            int nSlabs)
{
    __shared__ _Float16 WldsT[128 * WLDS_STRIDE];   // transposed: [n][k], zero-padded k>=K
    const int tid = threadIdx.x;
    for (int idx = tid; idx < 128 * 128; idx += blockDim.x) {
        int k = idx >> 7, n = idx & 127;
        float w = (k < K) ? W[k * 128 + n] : 0.0f;
        WldsT[n * WLDS_STRIDE + k] = (_Float16)w;
    }
    __syncthreads();

    const int lane = tid & 31;
    const int wave = tid >> 5;
    const int hi   = lane >> 4;       // lane half selects K range
    const int l15  = lane & 15;
    const int wavesTotal = gridDim.x * (blockDim.x >> 5);
    constexpr int kSteps = (K + 31) / 32;

    for (int slab = blockIdx.x * (blockDim.x >> 5) + wave; slab < nSlabs; slab += wavesTotal) {
        const int row0 = slab * 16;
        const float* arow = A + (size_t)(row0 + l15) * K;   // both lane halves: same 16 rows

        // prefetch next grid-stride slab's A row (global_prefetch_b8)
        if (slab + wavesTotal < nSlabs)
            __builtin_prefetch(A + (size_t)(slab + wavesTotal) * 16 * K + (size_t)l15 * K, 0, 3);

        v8f acc[8];
        #pragma unroll
        for (int t = 0; t < 8; ++t)
            #pragma unroll
            for (int r = 0; r < 8; ++r) acc[t][r] = 0.0f;

        #pragma unroll
        for (int ks = 0; ks < kSteps; ++ks) {
            const int k0 = ks * 32;
            // ---- A fragment (16x32 f16), ISA layout:
            //   VGPR j<4 -> K = 2j,2j+1 (+8 for hi lane half), j>=4 -> +16.
            v16h afrag;
            #pragma unroll
            for (int j = 0; j < 8; ++j) {
                const int kb = k0 + 2 * (j & 3) + ((j & 4) ? 16 : 0) + (hi ? 8 : 0);
                #pragma unroll
                for (int s = 0; s < 2; ++s) {
                    const int k = kb + s;
                    float v;
                    if constexpr (K % 32 == 0) {
                        v = arow[k];                       // unguarded, vectorizable
                    } else {
                        v = arow[(k < K) ? k : 0];         // clamped load + select, no branch
                        v = (k < K) ? v : 0.0f;
                    }
                    if constexpr (AFFINE) v = fmaf(v, scale[k & (K - 1)], shift[k & (K - 1)]);
                    afrag[2 * j + s] = (_Float16)v;
                }
            }
            // ---- 8 column tiles: B fragment (32x16 f16): same K layout as A (elem<8 -> K=8*hi+e, elem>=8 -> K=16+8*hi+e-8)
            #pragma unroll
            for (int nt = 0; nt < 8; ++nt) {
                const int n = nt * 16 + l15;
                const _Float16* wcol = &WldsT[n * WLDS_STRIDE + k0 + (hi ? 8 : 0)];
                v16h bfrag;
                #pragma unroll
                for (int j = 0; j < 8; ++j) {     // K-consecutive pairs -> ds_load_b128 runs
                    bfrag[2 * j]     = wcol[2 * j + ((j & 4) ? 8 : 0)];
                    bfrag[2 * j + 1] = wcol[2 * j + 1 + ((j & 4) ? 8 : 0)];
                }
                acc[nt] = __builtin_amdgcn_wmma_f32_16x16x32_f16(
                    false, afrag, false, bfrag, (short)0, acc[nt], false, false);
            }
        }
        // ---- epilogue: C/D layout: VGPR r -> row r + 8*hi, lanes give column
        #pragma unroll
        for (int nt = 0; nt < 8; ++nt) {
            const int col  = nt * 16 + l15;
            const float bc = bias[col];
            #pragma unroll
            for (int r = 0; r < 8; ++r) {
                const int row = row0 + r + hi * 8;
                const float v = acc[nt][r];
                lin[(size_t)row * GD + col] = v;
                const float di = dinv[row];
                agg[(size_t)row * GD + col] = di * di * v + bc;
            }
        }
    }
}

// ---------------------------------------------------------------- edge scatter
// agg[col] += dinv[row]*dinv[col] * lin[row]; 32 lanes per edge, float4 per lane.
// Edge index / dinv fetches forced wave-uniform -> scalar loads.
__global__ void k_edges(const long long* __restrict__ ei,
                        const float* __restrict__ lin,
                        const float* __restrict__ dinv,
                        float* __restrict__ agg, int ne)
{
    const int wave = __builtin_amdgcn_readfirstlane(threadIdx.x >> 5);
    const int e = blockIdx.x * 8 + wave;
    if (e >= ne) return;
    const int lane = threadIdx.x & 31;
    const int r = (int)ei[e];
    const int c = (int)ei[(size_t)ne + e];
    const float coef = dinv[r] * dinv[c];
    const float4 v = *(const float4*)(lin + (size_t)r * GD + lane * 4);
    float* dst = agg + (size_t)c * GD + lane * 4;
    atomicAddF(dst + 0, v.x * coef);
    atomicAddF(dst + 1, v.y * coef);
    atomicAddF(dst + 2, v.z * coef);
    atomicAddF(dst + 3, v.w * coef);
}

// ---------------------------------------------------------------- lrelu + BN stats
__global__ void k_stats(const float* __restrict__ agg, float* __restrict__ h,
                        float* __restrict__ sums, float* __restrict__ sumsq, int n)
{
    __shared__ float sm[256], sq[256];
    const int t = blockIdx.x * blockDim.x + threadIdx.x;
    const int f = t & 127;
    const int istep = (gridDim.x * blockDim.x) >> 7;
    float s = 0.0f, q = 0.0f;
    for (int i = t >> 7; i < n; i += istep) {
        const float v = agg[(size_t)i * GD + f];
        const float y = v > 0.0f ? v : 0.01f * v;
        h[(size_t)i * GD + f] = y;
        s += y; q += y * y;
    }
    sm[threadIdx.x] = s; sq[threadIdx.x] = q;
    __syncthreads();
    if (threadIdx.x < 128) {
        atomicAddF(&sums[f],  sm[threadIdx.x] + sm[threadIdx.x + 128]);
        atomicAddF(&sumsq[f], sq[threadIdx.x] + sq[threadIdx.x + 128]);
    }
}

__global__ void k_affine(const float* __restrict__ sums, const float* __restrict__ sumsq,
                         const float* __restrict__ g, const float* __restrict__ be,
                         float* __restrict__ scale, float* __restrict__ shift, float nInv)
{
    const int f = threadIdx.x;
    if (f < 128) {
        const float m  = sums[f] * nInv;
        const float vr = sumsq[f] * nInv - m * m;
        const float rs = rsqrtf(vr + BN_EPS) * g[f];
        scale[f] = rs;
        shift[f] = be[f] - m * rs;
    }
}

// ---------------------------------------------------------------- pooling + head
__global__ void k_pool(const float* __restrict__ agg, const long long* __restrict__ batch,
                       float* __restrict__ pool, float* __restrict__ cnt, int n)
{
    const int t = blockIdx.x * blockDim.x + threadIdx.x;
    const int f = t & 127;
    const int istep = (gridDim.x * blockDim.x) >> 7;
    for (int i = t >> 7; i < n; i += istep) {
        const float v = agg[(size_t)i * GD + f];
        const float y = v > 0.0f ? v : 0.01f * v;
        const int g = (int)batch[i];
        atomicAddF(&pool[(size_t)g * GD + f], y);
        if (f == 0) atomicAddF(&cnt[g], 1.0f);
    }
}

__global__ void k_final(const float* __restrict__ pool, const float* __restrict__ cnt,
                        const float* __restrict__ Wout, const float* __restrict__ bout,
                        float* __restrict__ out, int ng)
{
    const int wave = threadIdx.x >> 5;
    const int lane = threadIdx.x & 31;
    const int g = blockIdx.x * 8 + wave;
    if (g >= ng) return;
    const float4 p = *(const float4*)(pool + (size_t)g * GD + lane * 4);
    const float4 w = *(const float4*)(Wout + lane * 4);
    float partial = p.x * w.x + p.y * w.y + p.z * w.z + p.w * w.w;
    for (int off = 16; off; off >>= 1) partial += __shfl_down(partial, off, 32);
    if (lane == 0) out[g] = partial / fmaxf(cnt[g], 1.0f) + bout[0];
}

// ---------------------------------------------------------------- launch
extern "C" void kernel_launch(void* const* d_in, const int* in_sizes, int n_in,
                              void* d_out, int out_size, void* d_ws, size_t ws_size,
                              hipStream_t stream)
{
    const float*     x     = (const float*)d_in[0];
    const long long* ei    = (const long long*)d_in[1];
    const long long* batch = (const long long*)d_in[2];
    const float* Wl[4] = { (const float*)d_in[3], (const float*)d_in[5],
                           (const float*)d_in[7], (const float*)d_in[9] };
    const float* bl[4] = { (const float*)d_in[4], (const float*)d_in[6],
                           (const float*)d_in[8], (const float*)d_in[10] };
    const float* gl[3] = { (const float*)d_in[11], (const float*)d_in[13], (const float*)d_in[15] };
    const float* bel[3]= { (const float*)d_in[12], (const float*)d_in[14], (const float*)d_in[16] };
    const float* Wout  = (const float*)d_in[17];
    const float* bout  = (const float*)d_in[18];
    float* out = (float*)d_out;

    // workspace carve (floats)
    float* p = (float*)d_ws;
    float* lin   = p; p += (size_t)GN * GD;
    float* agg   = p; p += (size_t)GN * GD;
    float* h     = p; p += (size_t)GN * GD;
    float* deg   = p; p += GN;
    float* dinv  = p; p += GN;
    float* sums  = p; p += 128;
    float* sumsq = p; p += 128;
    float* scale = p; p += 128;
    float* shift = p; p += 128;
    float* pool  = p; p += (size_t)GG * GD;
    float* cnt   = p; p += GG;

    const int nSlabs = GN / 16;                    // 6250, exact
    const dim3 B256(256);
    const int gemmGrid  = 512;                     // grid-stride over slabs, 8 waves/block
    const int statsGrid = 512;
    const int edgeGrid  = (GE + 7) / 8;            // 8 edges per 256-thread block

    // degrees (self-loop included) -> dinv
    k_fill<<<(GN + 255) / 256, B256, 0, stream>>>(deg, 1.0f, GN);
    k_count_deg<<<(GE + 255) / 256, B256, 0, stream>>>(ei, deg, GE);
    k_dinv<<<(GN + 255) / 256, B256, 0, stream>>>(deg, dinv, GN);

    // ---- layer 0: K=9, no input affine
    k_gemm_wmma<GF_IN, false><<<gemmGrid, B256, 0, stream>>>(x, Wl[0], bl[0], scale, shift,
                                                             dinv, lin, agg, nSlabs);
    k_edges<<<edgeGrid, B256, 0, stream>>>(ei, lin, dinv, agg, GE);
    k_fill<<<1, B256, 0, stream>>>(sums, 0.0f, 256);           // sums+sumsq contiguous
    k_stats<<<statsGrid, B256, 0, stream>>>(agg, h, sums, sumsq, GN);
    k_affine<<<1, 128, 0, stream>>>(sums, sumsq, gl[0], bel[0], scale, shift, 1.0f / (float)GN);

    // ---- layers 1..3: K=128, BN folded into A-load
    for (int L = 1; L < 4; ++L) {
        k_gemm_wmma<GD, true><<<gemmGrid, B256, 0, stream>>>(h, Wl[L], bl[L], scale, shift,
                                                             dinv, lin, agg, nSlabs);
        k_edges<<<edgeGrid, B256, 0, stream>>>(ei, lin, dinv, agg, GE);
        if (L < 3) {
            k_fill<<<1, B256, 0, stream>>>(sums, 0.0f, 256);
            k_stats<<<statsGrid, B256, 0, stream>>>(agg, h, sums, sumsq, GN);
            k_affine<<<1, 128, 0, stream>>>(sums, sumsq, gl[L], bel[L], scale, shift,
                                            1.0f / (float)GN);
        }
    }

    // ---- global mean pool + head (lrelu fused into pooling)
    k_fill<<<(GG * GD + GG + 255) / 256, B256, 0, stream>>>(pool, 0.0f, GG * GD + GG);
    k_pool<<<statsGrid, B256, 0, stream>>>(agg, batch, pool, cnt, GN);
    k_final<<<GG / 8, B256, 0, stream>>>(pool, cnt, Wout, bout, out, GG);
    (void)in_sizes; (void)n_in; (void)out_size; (void)ws_size;
}
